// ConvIntrinsic_87093346828534
// MI455X (gfx1250) — matrix operand order repaired and checked by reading, not compiled, not run on hardware
//
#include <hip/hip_runtime.h>
#include <hip/hip_bf16.h>

// ---------------------------------------------------------------------------
// ConvIntrinsic fused as a single GEMM:
//   out[k, o*96+t] = relu( A[k, 0:1312] @ B[0:1312, o*96+t] + bias[t] )
// A[k, x*256+y*32+f] = barycentric interp (gathered from mesh_signal)
// A[k, 1280+f]       = mesh_signal[k, f]            (center contribution)
// B[(x,y,f),(o,t)]   = sum_{r,a} kernel[r,a,x,y] * W[t, r, (a+o)%8, f]
// B[1280+f,(o,t)]    = self[t, f]
// Precision: f16 operands, f32 accumulate via v_wmma_f32_16x16x32_f16.
// LDS tiles are double-buffered (1 barrier / k-step) and padded to an 80-byte
// row stride so every ds_load_b128 / ds_store is bank-conflict-free
// (dword lane stride 20; 20*n mod 64 distinct for n=0..15).
// ---------------------------------------------------------------------------

#define KTOT 100000
#define RDIM 5
#define ADIM 8
#define FDIM 32
#define TDIM 96
#define KD   1312      // R*A*F + F
#define NTOT 864       // 9 orientations * 96 templates
#define BM   128
#define BN   144
#define BK   32
#define LPAD 40        // padded LDS row stride in halves (80 B)
#define GRID_M 782     // ceil(100000/128)
#define GRID_N 6       // 864 / 144

typedef __attribute__((ext_vector_type(16))) _Float16 v16h;
typedef __attribute__((ext_vector_type(8)))  _Float16 v8h;
typedef __attribute__((ext_vector_type(8)))  float    v8f;

// ---------------------------------------------------------------------------
// Kernel 1: fold prior-kernel + rotated template weights + self weights into
// one f16 matrix Bt, COLUMN-major: Bt[n * 1312 + kk], n = o*96+t.
// 864*1312 elements, 40 MACs each — negligible, runs once per launch.
// ---------------------------------------------------------------------------
__global__ __launch_bounds__(256) void build_bt_kernel(
    const float* __restrict__ kern,   // (R,A,R,A)
    const float* __restrict__ Wn,     // (T,R,A,F)
    const float* __restrict__ Ws,     // (T,1,F)
    _Float16* __restrict__ Btg)       // (864,1312) f16
{
    int id = blockIdx.x * 256 + threadIdx.x;
    if (id >= NTOT * KD) return;
    int n  = id / KD;
    int kk = id - n * KD;
    int o  = n / TDIM;
    int t  = n - o * TDIM;
    float sum;
    if (kk < RDIM * ADIM * FDIM) {
        int x = kk >> 8;          // kk / 256
        int y = (kk >> 5) & 7;    // (kk / 32) % 8
        int f = kk & 31;
        sum = 0.f;
        #pragma unroll
        for (int rr = 0; rr < RDIM; ++rr) {
            #pragma unroll
            for (int aa = 0; aa < ADIM; ++aa) {
                float kv = kern[(((rr * ADIM + aa) * RDIM + x) * ADIM) + y];
                float wv = Wn[(((t * RDIM + rr) * ADIM + ((aa + o) & 7)) * FDIM) + f];
                sum += kv * wv;
            }
        }
    } else {
        sum = Ws[t * FDIM + (kk - RDIM * ADIM * FDIM)];
    }
    Btg[(size_t)n * KD + kk] = (_Float16)sum;
}

// ---------------------------------------------------------------------------
// Kernel 2: fused gather + GEMM + bias + ReLU.
// Block = 256 threads = 8 wave32s. Tile 128x144, BK=32 (one (x,y) cell/step).
// Wave w owns rows k0+16w..+15 and all 144 cols (9 WMMA fragments, 72 accVGPR).
// ---------------------------------------------------------------------------
__global__ __launch_bounds__(256) void conv_gemm_kernel(
    const float* __restrict__ mesh,   // (K,F)
    const float* __restrict__ bary,   // (K,R,A,3,2)
    const _Float16* __restrict__ Btg, // (864,1312) f16, column-major B
    const float* __restrict__ bias,   // (T,)
    float* __restrict__ out)          // (K,9,96)
{
    __shared__ alignas(16) _Float16 Atile[2][BM * LPAD];  // 2 x 10.0 KB
    __shared__ alignas(16) _Float16 Btile[2][BN * LPAD];  // 2 x 11.25 KB

    const int tid  = threadIdx.x;
    const int wave = tid >> 5;
    const int lane = tid & 31;
    const int k0   = blockIdx.x * BM;
    const int n0   = blockIdx.y * BN;

    v8f acc[9];
    #pragma unroll
    for (int i = 0; i < 9; ++i) acc[i] = (v8f){0.f,0.f,0.f,0.f,0.f,0.f,0.f,0.f};

    // A-tile build mapping: 2 threads per row, 16 f-values each
    const int r    = tid >> 1;      // 0..127
    const int half = tid & 1;       // 0 -> f 0..15, 1 -> f 16..31
    const int krow = k0 + r;

    // Build tiles for k-step `cell` into buffer `buf`.
    auto build_tiles = [&](int cell, int buf) {
        // ---- A tile: gather + barycentric interp, f32 -> f16 ----
        float s[16];
        #pragma unroll
        for (int q = 0; q < 16; ++q) s[q] = 0.f;
        if (krow < KTOT) {
            if (cell < 40) {
                const int x = cell >> 3;
                const int y = cell & 7;
                const float* bp = bary + ((((size_t)krow * RDIM + x) * ADIM + y) * 6);
                #pragma unroll
                for (int j = 0; j < 3; ++j) {
                    const int   ij = (int)bp[2 * j];
                    const float wj = bp[2 * j + 1];
                    const float4* mp =
                        (const float4*)(mesh + ((size_t)ij * FDIM + half * 16));
                    #pragma unroll
                    for (int qq = 0; qq < 4; ++qq) {
                        float4 mv = mp[qq];
                        s[qq*4+0] += wj * mv.x;  s[qq*4+1] += wj * mv.y;
                        s[qq*4+2] += wj * mv.z;  s[qq*4+3] += wj * mv.w;
                    }
                }
            } else {   // center columns: raw mesh signal
                const float4* mp =
                    (const float4*)(mesh + ((size_t)krow * FDIM + half * 16));
                #pragma unroll
                for (int qq = 0; qq < 4; ++qq) {
                    float4 mv = mp[qq];
                    s[qq*4+0] = mv.x;  s[qq*4+1] = mv.y;
                    s[qq*4+2] = mv.z;  s[qq*4+3] = mv.w;
                }
            }
        }
        v8h h0, h1;
        #pragma unroll
        for (int q = 0; q < 8; ++q) { h0[q] = (_Float16)s[q]; h1[q] = (_Float16)s[q+8]; }
        *(v8h*)&Atile[buf][r * LPAD + half * 16]     = h0;
        *(v8h*)&Atile[buf][r * LPAD + half * 16 + 8] = h1;

        // ---- B tile: 144 cols x 32 k (64 B per col, contiguous, L2-hot) ----
        if (tid < BN) {
            const int kk0 = cell * BK;
            const uint4* src = (const uint4*)(Btg + ((size_t)(n0 + tid) * KD + kk0));
            uint4* dst = (uint4*)&Btile[buf][tid * LPAD];
            #pragma unroll
            for (int q = 0; q < 4; ++q) dst[q] = src[q];
        }
    };

    build_tiles(0, 0);
    __syncthreads();

    for (int cell = 0; cell < 41; ++cell) {
        const int cur = cell & 1;
        if (cell < 40) build_tiles(cell + 1, cur ^ 1);   // prefetch next step

        // ---- fragments + WMMA (buffer `cur`) ----
        // A layout (16-bit A 16x32): lane<16 -> M=lane, K 0..7 & 16..23
        //                            lane>=16 -> M=lane-16, K 8..15 & 24..31
        const int arow  = wave * 16 + (lane & 15);
        const int abase = (lane >> 4) * 8;
        v8h alo = *(const v8h*)&Atile[cur][arow * LPAD + abase];
        v8h ahi = *(const v8h*)&Atile[cur][arow * LPAD + abase + 16];
        v16h afrag = __builtin_shufflevector(alo, ahi,
                        0,1,2,3,4,5,6,7,8,9,10,11,12,13,14,15);
        // B layout (32x16): lane<16 -> N=lane, K 0..7 & 16..23; lane>=16 -> K 8..15 & 24..31
        const int kb = (lane >> 4) * 8;
        #pragma unroll
        for (int fn = 0; fn < 9; ++fn) {
            const int bcol = fn * 16 + (lane & 15);
            v8h blo = *(const v8h*)&Btile[cur][bcol * LPAD + kb];
            v8h bhi = *(const v8h*)&Btile[cur][bcol * LPAD + kb + 16];
            v16h bfrag = __builtin_shufflevector(blo, bhi,
                            0,1,2,3,4,5,6,7,8,9,10,11,12,13,14,15);
            acc[fn] = __builtin_amdgcn_wmma_f32_16x16x32_f16(
                /*neg_a=*/false, afrag, /*neg_b=*/false, bfrag,
                /*c_mod=*/(short)0, acc[fn],
                /*reuse_a=*/false, /*reuse_b=*/false);
        }
        __syncthreads();   // next-step tiles complete; readers of `cur` done
    }

    // ---- epilogue: + bias[t], ReLU, store ----
    // D layout: VGPR v -> M=v (lanes 0-15) / M=v+8 (lanes 16-31); N = lane%16
    #pragma unroll
    for (int fn = 0; fn < 9; ++fn) {
        const int col = n0 + fn * 16 + (lane & 15);
        const float bv = bias[col % TDIM];
        #pragma unroll
        for (int v = 0; v < 8; ++v) {
            const int row = k0 + wave * 16 + ((lane < 16) ? v : v + 8);
            if (row < KTOT) {
                float val = acc[fn][v] + bv;
                out[(size_t)row * NTOT + col] = val > 0.f ? val : 0.f;
            }
        }
    }
}

extern "C" void kernel_launch(void* const* d_in, const int* in_sizes, int n_in,
                              void* d_out, int out_size, void* d_ws, size_t ws_size,
                              hipStream_t stream) {
    const float* mesh = (const float*)d_in[0];   // (K,F)
    const float* bary = (const float*)d_in[1];   // (K,R,A,3,2)
    const float* kern = (const float*)d_in[2];   // (R,A,R,A)
    const float* Wn   = (const float*)d_in[3];   // (T,R,A,F)
    const float* Ws   = (const float*)d_in[4];   // (T,1,F)
    const float* bias = (const float*)d_in[5];   // (T,)
    float* out        = (float*)d_out;           // (K,9,96) f32

    _Float16* Btg = (_Float16*)d_ws;             // 864*1312*2 B = 2.27 MB

    // 1) fold weights -> Bt (column-major f16), trivially cheap
    build_bt_kernel<<<(NTOT * KD + 255) / 256, 256, 0, stream>>>(kern, Wn, Ws, Btg);

    // 2) fused gather + GEMM + bias + ReLU
    conv_gemm_kernel<<<dim3(GRID_M, GRID_N), 256, 0, stream>>>(
        mesh, bary, Btg, bias, out);
}
